// GCNLSTM_39075612459407
// MI455X (gfx1250) — matrix operand order repaired and checked by reading, not compiled, not run on hardware
//
#include <hip/hip_runtime.h>
#include <hip/hip_bf16.h>

typedef _Float16 v16h __attribute__((ext_vector_type(16)));
typedef float    v8f  __attribute__((ext_vector_type(8)));

#define SLOPE 0.01f
#define BN_EPS 1e-5f

__device__ __forceinline__ float leakyf(float v) { return v >= 0.f ? v : SLOPE * v; }
__device__ __forceinline__ float sigmf(float v)  { return 1.f / (1.f + expf(-v)); }

// ---------------------------------------------------------------------------
// fp32-in / fp32-out GEMM via v_wmma_f32_16x16x32_f16.
// C[M,N] = A[M,K] @ Bt[N,K]^T   (torch Linear convention, no bias)
// REQUIRES: M%16==0, N%16==0, K%32==0, A/Bt 16B-aligned.
// One wave per 16x16 C tile. Fragments loaded as float4 (global_load_b128).
// ---------------------------------------------------------------------------
__device__ __forceinline__ void load_afrag(const float* __restrict__ ap, v16h& af) {
  // ap = arow + k0 + ka : halves 0..7 = ap[0..7], halves 8..15 = ap[16..23]
  float4 a0 = *(const float4*)(ap + 0);
  float4 a1 = *(const float4*)(ap + 4);
  float4 a2 = *(const float4*)(ap + 16);
  float4 a3 = *(const float4*)(ap + 20);
  af[0]  = (_Float16)a0.x; af[1]  = (_Float16)a0.y; af[2]  = (_Float16)a0.z; af[3]  = (_Float16)a0.w;
  af[4]  = (_Float16)a1.x; af[5]  = (_Float16)a1.y; af[6]  = (_Float16)a1.z; af[7]  = (_Float16)a1.w;
  af[8]  = (_Float16)a2.x; af[9]  = (_Float16)a2.y; af[10] = (_Float16)a2.z; af[11] = (_Float16)a2.w;
  af[12] = (_Float16)a3.x; af[13] = (_Float16)a3.y; af[14] = (_Float16)a3.z; af[15] = (_Float16)a3.w;
}

__device__ __forceinline__ void load_bfrag(const float* __restrict__ bp, v16h& bf) {
  // bp = brow + k0 + kb : halves 0..7 = bp[0..7], halves 8..15 = bp[16..23]
  float4 b0 = *(const float4*)(bp + 0);
  float4 b1 = *(const float4*)(bp + 4);
  float4 b2 = *(const float4*)(bp + 16);
  float4 b3 = *(const float4*)(bp + 20);
  bf[0]  = (_Float16)b0.x; bf[1]  = (_Float16)b0.y; bf[2]  = (_Float16)b0.z; bf[3]  = (_Float16)b0.w;
  bf[4]  = (_Float16)b1.x; bf[5]  = (_Float16)b1.y; bf[6]  = (_Float16)b1.z; bf[7]  = (_Float16)b1.w;
  bf[8]  = (_Float16)b2.x; bf[9]  = (_Float16)b2.y; bf[10] = (_Float16)b2.z; bf[11] = (_Float16)b2.w;
  bf[12] = (_Float16)b3.x; bf[13] = (_Float16)b3.y; bf[14] = (_Float16)b3.z; bf[15] = (_Float16)b3.w;
}

__global__ void gemm_nt_wmma(const float* __restrict__ A, const float* __restrict__ Bt,
                             float* __restrict__ C, int M, int N, int K) {
  int wave = (blockIdx.x * blockDim.x + threadIdx.x) >> 5;
  int lane = threadIdx.x & 31;
  int ntiles = N >> 4;
  int total = (M >> 4) * ntiles;
  if (wave >= total) return;
  int mt = wave / ntiles, nt = wave % ntiles;

  int m  = mt * 16 + (lane & 15);
  int n  = nt * 16 + (lane & 15);
  int ka = (lane >> 4) * 8;    // A-fragment K sub-offset
  int kb = (lane >> 4) * 8;    // B-fragment K sub-offset
  const float* arow = A + (size_t)m * K + ka;
  const float* brow = Bt + (size_t)n * K + kb;

  v8f acc = {};
  for (int k0 = 0; k0 < K; k0 += 32) {
    v16h af, bf;
    load_afrag(arow + k0, af);
    load_bfrag(brow + k0, bf);
    acc = __builtin_amdgcn_wmma_f32_16x16x32_f16(false, af, false, bf, (short)0, acc, false, false);
  }
  int mb = mt * 16 + (lane >> 4) * 8;
#pragma unroll
  for (int r = 0; r < 8; ++r) C[(size_t)(mb + r) * N + n] = acc[r];
}

// ---------------------------------------------------------------------------
// Split-K WMMA GEMM for the fat LSTM input projection (M=64, N=1024, K=64000).
// blockDim=128: 4 waves = 4 m-tiles sharing B reads (Wih0 streamed ~once).
// grid.x = (N/16) * kchunks. Accumulates into C with f32 atomics (C pre-zeroed).
// (K/kchunks) % 32 == 0.
// ---------------------------------------------------------------------------
__global__ void gemm_nt_wmma_splitk(const float* __restrict__ A, const float* __restrict__ Bt,
                                    float* __restrict__ C, int M, int N, int K, int kchunks) {
  int lane = threadIdx.x & 31;
  int mt   = threadIdx.x >> 5;
  if (mt * 16 >= M) return;
  int ntiles = N >> 4;
  int nt = blockIdx.x % ntiles;
  int kc = blockIdx.x / ntiles;
  int klen = K / kchunks;
  int kbeg = kc * klen, kend = kbeg + klen;

  int m  = mt * 16 + (lane & 15);
  int n  = nt * 16 + (lane & 15);
  int ka = (lane >> 4) * 8;
  int kb = (lane >> 4) * 8;
  const float* arow = A + (size_t)m * K + ka;
  const float* brow = Bt + (size_t)n * K + kb;

  v8f acc = {};
  for (int k0 = kbeg; k0 < kend; k0 += 32) {
    v16h af, bf;
    load_afrag(arow + k0, af);
    load_bfrag(brow + k0, bf);
    acc = __builtin_amdgcn_wmma_f32_16x16x32_f16(false, af, false, bf, (short)0, acc, false, false);
  }
  int mb = mt * 16 + (lane >> 4) * 8;
#pragma unroll
  for (int r = 0; r < 8; ++r) atomicAdd(&C[(size_t)(mb + r) * N + n], acc[r]);
}

// ---------------------------------------------------------------------------
// Small utility kernels
// ---------------------------------------------------------------------------
__global__ void fill_f32(float* p, float v, long n) {
  long i = blockIdx.x * (long)blockDim.x + threadIdx.x;
  if (i < n) p[i] = v;
}

// zero-pad rows: dst[M,Kdst] <- src[M,Ksrc] (Kdst > Ksrc, pad with 0)
__global__ void pad_k(const float* __restrict__ src, float* __restrict__ dst,
                      int M, int Ksrc, int Kdst) {
  long i = blockIdx.x * (long)blockDim.x + threadIdx.x;
  if (i >= (long)M * Kdst) return;
  int m = (int)(i / Kdst), k = (int)(i % Kdst);
  dst[i] = (k < Ksrc) ? src[(size_t)m * Ksrc + k] : 0.f;
}

__global__ void deg_count(const int* __restrict__ cols, float* __restrict__ deg, long E) {
  long i = blockIdx.x * (long)blockDim.x + threadIdx.x;
  if (i < E) atomicAdd(&deg[cols[i]], 1.0f);
}

__global__ void rsqrt_inplace(float* d, int n) {
  int i = blockIdx.x * blockDim.x + threadIdx.x;
  if (i < n) d[i] = rsqrtf(d[i]);   // deg >= 1 always (self-loop)
}

// GCN message scatter: agg[col] += hlin[row] * dinv[row]*dinv[col]
// CH/4 threads per edge, float4 gather, 4 f32 atomics (h fits in 192MB L2)
__global__ void edge_scatter(const int* __restrict__ rows, const int* __restrict__ cols,
                             const float* __restrict__ dinv, const float* __restrict__ hlin,
                             float* __restrict__ agg, long E, int CH) {
  int per = CH >> 2;
  long gid = blockIdx.x * (long)blockDim.x + threadIdx.x;
  long e = gid / per;
  if (e >= E) return;
  int c4 = (int)(gid % per) << 2;
  int r = rows[e], c = cols[e];
  float nrm = dinv[r] * dinv[c];
  const float4 v = *(const float4*)(hlin + (size_t)r * CH + c4);
  float* dst = agg + (size_t)c * CH + c4;
  atomicAdd(dst + 0, v.x * nrm);
  atomicAdd(dst + 1, v.y * nrm);
  atomicAdd(dst + 2, v.z * nrm);
  atomicAdd(dst + 3, v.w * nrm);
}

// self-loop contribution + bias: agg += hlin * dinv^2 + b
__global__ void self_bias(const float* __restrict__ hlin, const float* __restrict__ dinv,
                          const float* __restrict__ b, float* __restrict__ agg, int M, int CH) {
  long i = blockIdx.x * (long)blockDim.x + threadIdx.x;
  if (i >= (long)M * CH) return;
  int r = (int)(i / CH), c = (int)(i % CH);
  float di = dinv[r];
  agg[i] += hlin[i] * di * di + b[c];
}

// per-channel sum / sumsq (block = 128 channels x 128-row strip)
__global__ void bn_stats(const float* __restrict__ h, float* __restrict__ sums,
                         float* __restrict__ sumsq, int M, int CH) {
  int c  = threadIdx.x;
  int r0 = blockIdx.x * 128;
  int r1 = min(r0 + 128, M);
  float s = 0.f, q = 0.f;
  for (int r = r0; r < r1; ++r) {
    float v = h[(size_t)r * CH + c];
    s += v; q += v * v;
  }
  atomicAdd(&sums[c], s);
  atomicAdd(&sumsq[c], q);
}

// y = leaky(gamma*(x-mean)*rsqrt(var+eps)+beta); optional skip: y=(y+skip)/2
__global__ void bn_apply(const float* __restrict__ x, const float* __restrict__ sums,
                         const float* __restrict__ sumsq, const float* __restrict__ gamma,
                         const float* __restrict__ beta, const float* __restrict__ skip,
                         float* __restrict__ out, int M, int CH) {
  long i = blockIdx.x * (long)blockDim.x + threadIdx.x;
  if (i >= (long)M * CH) return;
  int c = (int)(i % CH);
  float inv_m = 1.f / (float)M;
  float mean = sums[c] * inv_m;
  float var  = sumsq[c] * inv_m - mean * mean;
  float y = gamma[c] * (x[i] - mean) * rsqrtf(var + BN_EPS) + beta[c];
  y = leakyf(y);
  if (skip) y = 0.5f * (y + skip[i]);
  out[i] = y;
}

__global__ void leaky_inplace(float* p, long n) {
  long i = blockIdx.x * (long)blockDim.x + threadIdx.x;
  if (i < n) p[i] = leakyf(p[i]);
}

// ---------------------------------------------------------------------------
// LSTM layer, sequential over T steps. One block, 1024 threads (32 wave32s).
// gates_pre: precomputed x@Wih^T [T,1024] (or null), else x projection inlined
// via xseq [T,xdim] @ Wih[1024,xdim]^T. Gate order: i,f,g,o (torch).
// Dot products use float4 (ds_load_b128 / global_load_b128).
// ---------------------------------------------------------------------------
__global__ void lstm_seq(const float* __restrict__ gates_pre,
                         const float* __restrict__ xseq, int xdim,
                         const float* __restrict__ Wih,
                         const float* __restrict__ Whh,
                         const float* __restrict__ bih, const float* __restrict__ bhh,
                         float* __restrict__ hs_out, float* __restrict__ h_last, int T) {
  const int HH = 256, H4 = 1024;
  __shared__ float sh_h[256], sh_c[256], sh_x[256], sh_g[1024];
  int tid = threadIdx.x;
  if (tid < HH) { sh_h[tid] = 0.f; sh_c[tid] = 0.f; }
  __syncthreads();
  for (int t = 0; t < T; ++t) {
    if (xdim > 0 && tid < xdim) sh_x[tid] = xseq[(size_t)t * xdim + tid];
    __syncthreads();
    float acc = bih[tid] + bhh[tid];
    if (gates_pre) acc += gates_pre[(size_t)t * H4 + tid];
    if (xdim > 0) {
      const float4* wr = (const float4*)(Wih + (size_t)tid * xdim);
      const float4* xx = (const float4*)sh_x;
      int n4 = xdim >> 2;
      for (int k = 0; k < n4; ++k) {
        float4 w = wr[k], v = xx[k];
        acc += w.x * v.x + w.y * v.y + w.z * v.z + w.w * v.w;
      }
    }
    {
      const float4* wh = (const float4*)(Whh + (size_t)tid * HH);
      const float4* hh = (const float4*)sh_h;
#pragma unroll 4
      for (int k = 0; k < HH / 4; ++k) {
        float4 w = wh[k], v = hh[k];
        acc += w.x * v.x + w.y * v.y + w.z * v.z + w.w * v.w;
      }
    }
    sh_g[tid] = acc;
    __syncthreads();
    if (tid < HH) {
      float i_ = sigmf(sh_g[tid]);
      float f_ = sigmf(sh_g[HH + tid]);
      float g_ = tanhf(sh_g[2 * HH + tid]);
      float o_ = sigmf(sh_g[3 * HH + tid]);
      float cn = f_ * sh_c[tid] + i_ * g_;
      sh_c[tid] = cn;
      float hn = o_ * tanhf(cn);
      sh_h[tid] = hn;
      if (hs_out) hs_out[(size_t)t * HH + tid] = hn;
    }
    __syncthreads();
  }
  if (h_last && tid < HH) h_last[tid] = sh_h[tid];
}

// regression head: leaky(h) -> leaky(Wr1) -> leaky(Wrm) -> Wend. One block.
__global__ void reghead(const float* __restrict__ h_last,
                        const float* __restrict__ Wr1, const float* __restrict__ br1,
                        const float* __restrict__ Wrm, const float* __restrict__ brm,
                        const float* __restrict__ Wend, const float* __restrict__ bend,
                        float* __restrict__ out) {
  __shared__ float v0[256], v1[128], v2[128];
  int tid = threadIdx.x;
  if (tid < 256) v0[tid] = leakyf(h_last[tid]);
  __syncthreads();
  if (tid < 128) {
    float a = br1[tid];
    const float* w = Wr1 + (size_t)tid * 256;
    for (int k = 0; k < 256; ++k) a += w[k] * v0[k];
    v1[tid] = leakyf(a);
  }
  __syncthreads();
  if (tid < 128) {
    float a = brm[tid];
    const float* w = Wrm + (size_t)tid * 128;
    for (int k = 0; k < 128; ++k) a += w[k] * v1[k];
    v2[tid] = leakyf(a);
  }
  __syncthreads();
  if (tid == 0) {
    float a = bend[0];
    for (int k = 0; k < 128; ++k) a += Wend[k] * v2[k];
    out[0] = a;
  }
}

// ---------------------------------------------------------------------------
static inline unsigned nblk(long n, int t) { return (unsigned)((n + t - 1) / t); }

extern "C" void kernel_launch(void* const* d_in, const int* in_sizes, int n_in,
                              void* d_out, int out_size, void* d_ws, size_t ws_size,
                              hipStream_t stream) {
  const int   NN = 128000;      // total nodes
  const long  E  = 2048000;
  const int   H  = 128, C = 32, T = 64, H4 = 1024;

  const float* x    = (const float*)d_in[0];
  const int*   ei   = (const int*)d_in[1];
  const int*   rows = ei;            // edge_index[0] = source
  const int*   cols = ei + E;        // edge_index[1] = target
  const float* W1   = (const float*)d_in[2];
  const float* b1   = (const float*)d_in[3];
  const float* Wm   = (const float*)d_in[4];   // [1,128,128]
  const float* bm   = (const float*)d_in[5];
  const float* Wf   = (const float*)d_in[6];
  const float* bf   = (const float*)d_in[7];
  const float* bn_g = (const float*)d_in[8];
  const float* bn_b = (const float*)d_in[9];
  const float* Wih0 = (const float*)d_in[10];
  const float* Whh0 = (const float*)d_in[11];
  const float* bih0 = (const float*)d_in[12];
  const float* bhh0 = (const float*)d_in[13];
  const float* Wih1 = (const float*)d_in[14];
  const float* Whh1 = (const float*)d_in[15];
  const float* bih1 = (const float*)d_in[16];
  const float* bhh1 = (const float*)d_in[17];
  const float* Wr1  = (const float*)d_in[18];
  const float* br1  = (const float*)d_in[19];
  const float* Wrm  = (const float*)d_in[20];
  const float* brm  = (const float*)d_in[21];
  const float* Wend = (const float*)d_in[22];
  const float* bend = (const float*)d_in[23];

  // workspace layout (floats)
  float* ws     = (float*)d_ws;
  float* dinv   = ws;                                 // 128000 (deg -> dinv in place)
  float* bufA   = dinv + NN;                          // 128000*128
  float* bufB   = bufA + (size_t)NN * H;              // 128000*128
  float* bufC   = bufB + (size_t)NN * H;              // 128000*128
  float* sums   = bufC + (size_t)NN * H;              // 128
  float* sumsq  = sums + H;                           // 128
  float* gates0 = sumsq + H;                          // 64*1024
  float* hs0    = gates0 + (size_t)T * H4;            // 64*256
  float* hlast  = hs0 + (size_t)T * 256;              // 256
  float* w1pad  = hlast + 256;                        // 128*32

  const long MH = (long)NN * H;                       // 16.384M
  const long MC = (long)NN * C;                       // 4.096M

  // ---- degree / symmetric normalization -------------------------------
  fill_f32<<<nblk(NN, 256), 256, 0, stream>>>(dinv, 1.0f, NN);          // self-loops
  deg_count<<<nblk(E, 256), 256, 0, stream>>>(cols, dinv, E);
  rsqrt_inplace<<<nblk(NN, 256), 256, 0, stream>>>(dinv, NN);

  // ---- GCN layer 1: pad K 16->32, x@W1^T -> bn -> leaky ----------------
  pad_k<<<nblk((long)NN * 32, 256), 256, 0, stream>>>(x, bufC, NN, 16, 32);
  pad_k<<<nblk(128 * 32, 256), 256, 0, stream>>>(W1, w1pad, H, 16, 32);
  {
    long waves = (long)(NN / 16) * (H / 16);
    gemm_nt_wmma<<<nblk(waves * 32, 256), 256, 0, stream>>>(bufC, w1pad, bufA, NN, H, 32);
  }
  fill_f32<<<nblk(MH, 256), 256, 0, stream>>>(bufB, 0.f, MH);
  edge_scatter<<<nblk(E * (H / 4), 256), 256, 0, stream>>>(rows, cols, dinv, bufA, bufB, E, H);
  self_bias<<<nblk(MH, 256), 256, 0, stream>>>(bufA, dinv, b1, bufB, NN, H);
  fill_f32<<<1, 256, 0, stream>>>(sums, 0.f, 2 * H);
  bn_stats<<<nblk(NN, 128), H, 0, stream>>>(bufB, sums, sumsq, NN, H);
  bn_apply<<<nblk(MH, 256), 256, 0, stream>>>(bufB, sums, sumsq, bn_g, bn_b, nullptr, bufB, NN, H);
  // h1 in bufB

  // ---- GCN mid layer: bn -> leaky -> skip avg -------------------------
  {
    long waves = (long)(NN / 16) * (H / 16);
    gemm_nt_wmma<<<nblk(waves * 32, 256), 256, 0, stream>>>(bufB, Wm, bufA, NN, H, H);
  }
  fill_f32<<<nblk(MH, 256), 256, 0, stream>>>(bufC, 0.f, MH);
  edge_scatter<<<nblk(E * (H / 4), 256), 256, 0, stream>>>(rows, cols, dinv, bufA, bufC, E, H);
  self_bias<<<nblk(MH, 256), 256, 0, stream>>>(bufA, dinv, bm, bufC, NN, H);
  fill_f32<<<1, 256, 0, stream>>>(sums, 0.f, 2 * H);
  bn_stats<<<nblk(NN, 128), H, 0, stream>>>(bufC, sums, sumsq, NN, H);
  bn_apply<<<nblk(MH, 256), 256, 0, stream>>>(bufC, sums, sumsq, bn_g, bn_b, bufB, bufC, NN, H);
  // h2 in bufC

  // ---- GCN final layer: -> [NN, 32], leaky only -----------------------
  {
    long waves = (long)(NN / 16) * (C / 16);
    gemm_nt_wmma<<<nblk(waves * 32, 256), 256, 0, stream>>>(bufC, Wf, bufA, NN, C, H);
  }
  fill_f32<<<nblk(MC, 256), 256, 0, stream>>>(bufB, 0.f, MC);
  edge_scatter<<<nblk(E * (C / 4), 256), 256, 0, stream>>>(rows, cols, dinv, bufA, bufB, E, C);
  self_bias<<<nblk(MC, 256), 256, 0, stream>>>(bufA, dinv, bf, bufB, NN, C);
  leaky_inplace<<<nblk(MC, 256), 256, 0, stream>>>(bufB, MC);
  // seq = bufB viewed as [64, 64000] (reshape is flat-contiguous)

  // ---- LSTM layer-0 input projection: [64,64000]@Wih0^T (split-K) -----
  fill_f32<<<nblk((long)T * H4, 256), 256, 0, stream>>>(gates0, 0.f, (long)T * H4);
  {
    const int KCH = 16;  // 64000/16 = 4000 (multiple of 32)
    gemm_nt_wmma_splitk<<<(H4 / 16) * KCH, 128, 0, stream>>>(bufB, Wih0, gates0,
                                                             T, H4, 64000, KCH);
  }

  // ---- LSTM recurrences -----------------------------------------------
  lstm_seq<<<1, 1024, 0, stream>>>(gates0, nullptr, 0, nullptr, Whh0, bih0, bhh0,
                                   hs0, nullptr, T);
  lstm_seq<<<1, 1024, 0, stream>>>(nullptr, hs0, 256, Wih1, Whh1, bih1, bhh1,
                                   nullptr, hlast, T);

  // ---- regression head -------------------------------------------------
  reghead<<<1, 256, 0, stream>>>(hlast, Wr1, br1, Wrm, brm, Wend, bend, (float*)d_out);
}
